// DCAttention_78520592106188
// MI455X (gfx1250) — compile-verified, matrix operand order checked by reading, not run on hardware
//
#include <hip/hip_runtime.h>
#include <math.h>

typedef __attribute__((ext_vector_type(16))) __bf16 v16bf;
typedef __attribute__((ext_vector_type(8)))  __bf16 v8bf;
typedef __attribute__((ext_vector_type(8)))  float  v8f;

constexpr int Bsz = 2, Lseq = 2048, Dm = 1024, Hn = 16, DKh = 64;
constexpr int M_ROWS = Bsz * Lseq;   // 4096

static __device__ __forceinline__ v16bf mk16(v8bf lo, v8bf hi) {
    return __builtin_shufflevector(lo, hi, 0,1,2,3,4,5,6,7,8,9,10,11,12,13,14,15);
}

// async memory->LDS copy, 16B per lane (ASYNCcnt tracked)
static __device__ __forceinline__ void async_b128(unsigned lds_off, const void* gptr) {
    asm volatile("global_load_async_to_lds_b128 %0, %1, off"
                 :: "v"(lds_off), "v"((unsigned long long)(uintptr_t)gptr)
                 : "memory");
}
static __device__ __forceinline__ void wait_async0() {
    asm volatile("s_wait_asynccnt 0x0" ::: "memory");
}

// ---------------------------------------------------------------- utility ---
__global__ void cvt_f32_bf16(const float* __restrict__ s, __bf16* __restrict__ d, int n) {
    int i = blockIdx.x * 256 + threadIdx.x;
    if (i < n) d[i] = (__bf16)s[i];
}

// conv weights (O=D, I=D, K=3) fp32  ->  (N=D, 3*D) bf16 with kc = k*D + ci
__global__ void repack_conv_w(const float* __restrict__ s, __bf16* __restrict__ d) {
    int i = blockIdx.x * 256 + threadIdx.x;        // over D*3D
    int n  = i / (3 * Dm);
    int kc = i % (3 * Dm);
    int k  = kc / Dm, ci = kc % Dm;
    d[i] = (__bf16)s[(n * Dm + ci) * 3 + k];
}

// copy Qcat[:, :D] (ldc=2D) into per-batch padded buffer (B, L+2, D), interior rows
__global__ void pad_copy(const __bf16* __restrict__ cat, __bf16* __restrict__ pad) {
    int i = blockIdx.x * 256 + threadIdx.x;        // over B*L*D
    int d = i % Dm;
    int l = (i / Dm) % Lseq;
    int b = i / (Dm * Lseq);
    pad[((size_t)b * (Lseq + 2) + l + 1) * Dm + d] = cat[((size_t)b * Lseq + l) * 2 * Dm + d];
}

__global__ void pad_zero(__bf16* __restrict__ pad) {
    int i = blockIdx.x * 256 + threadIdx.x;        // over B*2*D
    if (i >= Bsz * 2 * Dm) return;
    int d = i % Dm;
    int r = (i / Dm) & 1;
    int b = i / (2 * Dm);
    pad[((size_t)b * (Lseq + 2) + (r ? (Lseq + 1) : 0)) * Dm + d] = (__bf16)0.0f;
}

// ------------------------------------------------------------- WMMA GEMM ---
// C(M,N) = A_mapped(M,KTOT) * Bw(N,KTOT)^T + bias.
// A row mapping: arow = (m>>LRUN_LG)*LSTRIDE + (m & mask) + kbase/KSPLIT  ->
// the same kernel does the 3-tap conv as a K=3D GEMM over per-batch padded rows.
// Staging: global_load_async_to_lds_b128 into double-buffered LDS, overlapped
// with the WMMA pipe; per-wave s_wait_asynccnt + block barrier.
template<int KTOT, int KSPLIT, int LRUN_LG, int LSTRIDE, int LDA, int LDC>
__global__ __launch_bounds__(256) void gemm_wmma(
    const __bf16* __restrict__ A, const __bf16* __restrict__ Bw,
    const float* __restrict__ bias, __bf16* __restrict__ Cb, float* __restrict__ Cf)
{
    constexpr int BM = 128, BN = 128, BK = 32;
    constexpr int NK = KTOT / BK;
    constexpr unsigned BUF_BYTES = 16384;   // 8KB A + 8KB B per buffer
    extern __shared__ __align__(16) char smem[];   // 2 * BUF_BYTES

    const int tid  = threadIdx.x;
    const int bm   = blockIdx.y * BM, bn = blockIdx.x * BN;
    const int w    = tid >> 5, lane = tid & 31;
    const int lm   = lane & 15;
    const int kh   = (lane & 16) ? 8 : 0;
    const int wm   = (w & 3) * 32, wn = (w >> 2) * 64;

    v8f acc[2][4];
#pragma unroll
    for (int i = 0; i < 2; i++)
#pragma unroll
        for (int j = 0; j < 4; j++) acc[i][j] = (v8f){};

    const int sr = tid >> 1;           // staging row 0..127
    const int sc = (tid & 1) * 16;     // staging col 0 / 16

    auto stage = [&](int kt, int buf) {
        const int kbase = kt * BK;
        const int shift = kbase / KSPLIT;
        const int acol  = kbase % KSPLIT + sc;
        const int m     = bm + sr;
        const int arow  = (m >> LRUN_LG) * LSTRIDE + (m & ((1 << LRUN_LG) - 1)) + shift;
        const __bf16* ga = A  + (size_t)arow * LDA + acol;
        const __bf16* gb = Bw + (size_t)(bn + sr) * KTOT + kbase + sc;
        const unsigned la = (unsigned)buf * BUF_BYTES + (unsigned)tid * 32u;  // A slice
        const unsigned lb = la + 8192u;                                        // B slice
        async_b128(la,       ga);
        async_b128(la + 16u, (const char*)ga + 16);
        async_b128(lb,       gb);
        async_b128(lb + 16u, (const char*)gb + 16);
    };

    stage(0, 0);
    wait_async0();
    __syncthreads();

    for (int kt = 0; kt < NK; kt++) {
        const int cur = kt & 1;
        if (kt + 1 < NK) stage(kt + 1, (kt + 1) & 1);   // DMA next tile (overlapped)

        const __bf16* sAc = (const __bf16*)(smem + (size_t)cur * BUF_BYTES);
        const __bf16* sBc = sAc + 4096;                  // +8KB

        v16bf af[2], bf[4];
#pragma unroll
        for (int i = 0; i < 2; i++) {
            int r = wm + i * 16 + lm;
            af[i] = mk16(*(const v8bf*)(sAc + r * 32 + kh),
                         *(const v8bf*)(sAc + r * 32 + kh + 16));
        }
#pragma unroll
        for (int j = 0; j < 4; j++) {
            int r = wn + j * 16 + lm;
            bf[j] = mk16(*(const v8bf*)(sBc + r * 32 + kh),
                         *(const v8bf*)(sBc + r * 32 + kh + 16));
        }
#pragma unroll
        for (int i = 0; i < 2; i++)
#pragma unroll
            for (int j = 0; j < 4; j++)
                acc[i][j] = __builtin_amdgcn_wmma_f32_16x16x32_bf16(
                    false, af[i], false, bf[j], (short)0, acc[i][j], false, false);

        wait_async0();       // this wave's DMA into the next buffer has landed
        __syncthreads();     // everyone done reading cur + all DMAs visible
    }

    const int half = lane >> 4;
#pragma unroll
    for (int i = 0; i < 2; i++) {
        int mrow = bm + wm + i * 16 + half * 8;
#pragma unroll
        for (int j = 0; j < 4; j++) {
            int   ncol = bn + wn + j * 16 + lm;
            float bv   = bias[ncol];
#pragma unroll
            for (int r = 0; r < 8; r++) {
                float  v   = acc[i][j][r] + bv;
                size_t idx = (size_t)(mrow + r) * LDC + ncol;
                if (Cb) Cb[idx] = (__bf16)v;
                if (Cf) Cf[idx] = v;
            }
        }
    }
}

// -------------------------------------------------- tau / delta small path ---
__global__ void dw_gelu(const float* __restrict__ x, const float* __restrict__ w1,
                        const float* __restrict__ b1, __bf16* __restrict__ g)
{
    int i  = blockIdx.x * 256 + threadIdx.x;       // over B*L*2D
    int o  = i % (2 * Dm);
    int l  = (i / (2 * Dm)) % Lseq;
    int b  = i / (2 * Dm * Lseq);
    int ci = o >> 1;
    float acc = b1[o];
#pragma unroll
    for (int k = 0; k < 3; k++) {
        int l2 = l + k - 1;
        if (l2 >= 0 && l2 < Lseq)
            acc += w1[o * 3 + k] * x[((size_t)b * Lseq + l2) * Dm + ci];
    }
    float ge = 0.5f * acc * (1.0f + erff(acc * 0.70710678118f));
    g[i] = (__bf16)ge;
}

__global__ void proj_sig(const __bf16* __restrict__ g, const float* __restrict__ w2,
                         const float* __restrict__ b2, float* __restrict__ out)
{
    int i = blockIdx.x * 256 + threadIdx.x;        // over B*L*H
    int h = i % Hn;
    int l = (i / Hn) % Lseq;
    int b = i / (Hn * Lseq);
    const __bf16* gr = g  + ((size_t)b * Lseq + l) * 2 * Dm;
    const float*  wr = w2 + (size_t)h * 2 * Dm;
    float acc = b2[h];
    for (int o = 0; o < 2 * Dm; o++) acc += wr[o] * (float)gr[o];
    out[((size_t)b * Hn + h) * Lseq + l] = 1.0f / (1.0f + __expf(-acc));
}

// ------------------------------------------------------- flash attention ---
__global__ __launch_bounds__(256) void flash_attn(
    const __bf16* __restrict__ Q, const __bf16* __restrict__ K,
    const __bf16* __restrict__ V, const float* __restrict__ tau,
    const float* __restrict__ delta, __bf16* __restrict__ O)
{
    __shared__ __align__(16) __bf16 sV[64][64];          // V tile transposed [ch][k]
    __shared__ __align__(16) __bf16 sP[8][16][64];       // per-wave P tile

    const int b = blockIdx.z, h = blockIdx.y;
    const int tid = threadIdx.x, w = tid >> 5, lane = tid & 31;
    const int lm = lane & 15, half = lane >> 4;
    const int kh = half ? 8 : 0;
    const int qbase = blockIdx.x * 128 + w * 16;
    const int hc0   = h * DKh;

    const __bf16* qrow = Q + ((size_t)(b * Lseq) + qbase + lm) * Dm + hc0;
    v16bf qf[2];
#pragma unroll
    for (int kc = 0; kc < 2; kc++)
        qf[kc] = mk16(*(const v8bf*)(qrow + kc * 32 + kh),
                      *(const v8bf*)(qrow + kc * 32 + kh + 16));

    const float* taup = tau   + ((size_t)b * Hn + h) * Lseq;
    const float* delp = delta + ((size_t)b * Hn + h) * Lseq;
    float taur[8];
#pragma unroll
    for (int r = 0; r < 8; r++) taur[r] = taup[qbase + half * 8 + r] * 0.125f; // 1/sqrt(64)

    float mrow[8], lrow[8];
    v8f   of[4];
#pragma unroll
    for (int r = 0; r < 8; r++) { mrow[r] = -3.0e38f; lrow[r] = 0.0f; }
#pragma unroll
    for (int j = 0; j < 4; j++) of[j] = (v8f){};

    const int vt_kr = tid >> 2, vt_cs = (tid & 3) * 16;

    for (int kt = 0; kt < Lseq / 64; kt++) {
        // warm L2 / WGP$ for next K/V tiles (global_prefetch_b8)
        if (kt + 1 < Lseq / 64) {
            const size_t nb = (size_t)(b * Lseq) + (kt + 1) * 64;
            __builtin_prefetch(K + (nb + lane)      * Dm + hc0, 0, 1);
            __builtin_prefetch(K + (nb + 32 + lane) * Dm + hc0, 0, 1);
            __builtin_prefetch(V + (nb + lane)      * Dm + hc0, 0, 1);
            __builtin_prefetch(V + (nb + 32 + lane) * Dm + hc0, 0, 1);
        }

        __syncthreads();
        {   // stage V tile transposed into LDS
            const __bf16* vr = V + ((size_t)(b * Lseq) + kt * 64 + vt_kr) * Dm + hc0 + vt_cs;
#pragma unroll
            for (int e = 0; e < 16; e++) sV[vt_cs + e][vt_kr] = vr[e];
        }
        __syncthreads();

        // S = Q K^T  (K fragments straight from global: channels contiguous)
        v8f sf[4];
#pragma unroll
        for (int j = 0; j < 4; j++) {
            const __bf16* krow = K + ((size_t)(b * Lseq) + kt * 64 + j * 16 + lm) * Dm + hc0;
            v8f a = (v8f){};
#pragma unroll
            for (int kc = 0; kc < 2; kc++) {
                v16bf kf = mk16(*(const v8bf*)(krow + kc * 32 + kh),
                                *(const v8bf*)(krow + kc * 32 + kh + 16));
                a = __builtin_amdgcn_wmma_f32_16x16x32_bf16(
                        false, qf[kc], false, kf, (short)0, a, false, false);
            }
            sf[j] = a;
        }

        // scale by tau/sqrt(DK) per q-row, add delta per key column
#pragma unroll
        for (int j = 0; j < 4; j++) {
            float dv = delp[kt * 64 + j * 16 + lm];
#pragma unroll
            for (int r = 0; r < 8; r++) sf[j][r] = sf[j][r] * taur[r] + dv;
        }

        // online softmax over the 64 key columns
        float rmax[8];
#pragma unroll
        for (int r = 0; r < 8; r++) {
            float v = fmaxf(fmaxf(sf[0][r], sf[1][r]), fmaxf(sf[2][r], sf[3][r]));
#pragma unroll
            for (int off = 1; off < 16; off <<= 1)
                v = fmaxf(v, __shfl_xor(v, off, 32));
            rmax[r] = v;
        }

        float corr[8], rsum[8];
#pragma unroll
        for (int r = 0; r < 8; r++) {
            float mn = fmaxf(mrow[r], rmax[r]);
            corr[r] = __expf(mrow[r] - mn);
            mrow[r] = mn;
            rsum[r] = 0.0f;
        }
#pragma unroll
        for (int j = 0; j < 4; j++)
#pragma unroll
            for (int r = 0; r < 8; r++) {
                float p = __expf(sf[j][r] - mrow[r]);
                sf[j][r] = p;
                rsum[r] += p;
            }
#pragma unroll
        for (int r = 0; r < 8; r++) {
#pragma unroll
            for (int off = 1; off < 16; off <<= 1)
                rsum[r] += __shfl_xor(rsum[r], off, 32);
            lrow[r] = lrow[r] * corr[r] + rsum[r];
        }
#pragma unroll
        for (int j = 0; j < 4; j++)
#pragma unroll
            for (int r = 0; r < 8; r++) of[j][r] *= corr[r];

        // C-layout P -> A-layout bf16 via per-wave LDS round trip (in-order per wave)
#pragma unroll
        for (int j = 0; j < 4; j++)
#pragma unroll
            for (int r = 0; r < 8; r++)
                sP[w][half * 8 + r][j * 16 + lm] = (__bf16)sf[j][r];

        v16bf pf[2];
#pragma unroll
        for (int kc = 0; kc < 2; kc++)
            pf[kc] = mk16(*(const v8bf*)&sP[w][lm][kc * 32 + kh],
                          *(const v8bf*)&sP[w][lm][kc * 32 + kh + 16]);

        // O += P V
#pragma unroll
        for (int j = 0; j < 4; j++)
#pragma unroll
            for (int kc = 0; kc < 2; kc++) {
                v16bf vf = mk16(*(const v8bf*)&sV[j * 16 + lm][kc * 32 + kh],
                                *(const v8bf*)&sV[j * 16 + lm][kc * 32 + kh + 16]);
                of[j] = __builtin_amdgcn_wmma_f32_16x16x32_bf16(
                            false, pf[kc], false, vf, (short)0, of[j], false, false);
            }
    }

#pragma unroll
    for (int j = 0; j < 4; j++)
#pragma unroll
        for (int r = 0; r < 8; r++) {
            float v = of[j][r] / lrow[r];
            O[((size_t)(b * Lseq) + qbase + half * 8 + r) * Dm + hc0 + j * 16 + lm] = (__bf16)v;
        }
}

// ------------------------------------------------------------------ host ---
extern "C" void kernel_launch(void* const* d_in, const int* in_sizes, int n_in,
                              void* d_out, int out_size, void* d_ws, size_t ws_size,
                              hipStream_t stream)
{
    const float* x       = (const float*)d_in[0];
    const float* Wq_w    = (const float*)d_in[1];
    const float* Wq_b    = (const float*)d_in[2];
    const float* Wk_w    = (const float*)d_in[3];
    const float* Wk_b    = (const float*)d_in[4];
    const float* Wv_w    = (const float*)d_in[5];
    const float* Wv_b    = (const float*)d_in[6];
    const float* convq_w = (const float*)d_in[7];
    const float* convq_b = (const float*)d_in[8];
    const float* convk_w = (const float*)d_in[9];
    const float* convk_b = (const float*)d_in[10];
    const float* qproj_w = (const float*)d_in[11];
    const float* qproj_b = (const float*)d_in[12];
    const float* kproj_w = (const float*)d_in[13];
    const float* kproj_b = (const float*)d_in[14];
    const float* tau1_w  = (const float*)d_in[15];
    const float* tau1_b  = (const float*)d_in[16];
    const float* tau2_w  = (const float*)d_in[17];
    const float* tau2_b  = (const float*)d_in[18];
    const float* del1_w  = (const float*)d_in[19];
    const float* del1_b  = (const float*)d_in[20];
    const float* del2_w  = (const float*)d_in[21];
    const float* del2_b  = (const float*)d_in[22];
    const float* out_w   = (const float*)d_in[23];
    const float* out_b   = (const float*)d_in[24];
    float* out = (float*)d_out;
    (void)in_sizes; (void)n_in; (void)out_size; (void)ws_size;

    char*  ws  = (char*)d_ws;
    size_t off = 0;
    auto alloc = [&](size_t bytes) -> char* {
        char* p = ws + off;
        off = (off + bytes + 255) & ~(size_t)255;
        return p;
    };

    __bf16* xb      = (__bf16*)alloc((size_t)M_ROWS * Dm * 2);
    __bf16* WqB     = (__bf16*)alloc((size_t)Dm * Dm * 2);
    __bf16* WkB     = (__bf16*)alloc((size_t)Dm * Dm * 2);
    __bf16* WvB     = (__bf16*)alloc((size_t)Dm * Dm * 2);
    __bf16* qprojB  = (__bf16*)alloc((size_t)Dm * 2 * Dm * 2);
    __bf16* kprojB  = (__bf16*)alloc((size_t)Dm * 2 * Dm * 2);
    __bf16* outwB   = (__bf16*)alloc((size_t)Dm * Dm * 2);
    __bf16* convqR  = (__bf16*)alloc((size_t)Dm * 3 * Dm * 2);
    __bf16* convkR  = (__bf16*)alloc((size_t)Dm * 3 * Dm * 2);
    __bf16* Qpad    = (__bf16*)alloc((size_t)Bsz * (Lseq + 2) * Dm * 2);
    __bf16* Kpad    = (__bf16*)alloc((size_t)Bsz * (Lseq + 2) * Dm * 2);
    __bf16* Qcat    = (__bf16*)alloc((size_t)M_ROWS * 2 * Dm * 2);
    __bf16* Kcat    = (__bf16*)alloc((size_t)M_ROWS * 2 * Dm * 2);
    __bf16* Qbuf    = (__bf16*)alloc((size_t)M_ROWS * Dm * 2);
    __bf16* Kbuf    = (__bf16*)alloc((size_t)M_ROWS * Dm * 2);
    __bf16* Vbuf    = (__bf16*)alloc((size_t)M_ROWS * Dm * 2);
    __bf16* gbuf    = (__bf16*)alloc((size_t)M_ROWS * 2 * Dm * 2);
    float*  taubuf  = (float*)alloc((size_t)Bsz * Hn * Lseq * 4);
    float*  delbuf  = (float*)alloc((size_t)Bsz * Hn * Lseq * 4);
    __bf16* attnO   = (__bf16*)alloc((size_t)M_ROWS * Dm * 2);

    auto cvt = [&](const float* s, __bf16* d, int n) {
        cvt_f32_bf16<<<(n + 255) / 256, 256, 0, stream>>>(s, d, n);
    };
    cvt(x, xb, M_ROWS * Dm);
    cvt(Wq_w, WqB, Dm * Dm);
    cvt(Wk_w, WkB, Dm * Dm);
    cvt(Wv_w, WvB, Dm * Dm);
    cvt(qproj_w, qprojB, Dm * 2 * Dm);
    cvt(kproj_w, kprojB, Dm * 2 * Dm);
    cvt(out_w, outwB, Dm * Dm);
    repack_conv_w<<<(Dm * 3 * Dm) / 256, 256, 0, stream>>>(convq_w, convqR);
    repack_conv_w<<<(Dm * 3 * Dm) / 256, 256, 0, stream>>>(convk_w, convkR);

    const dim3 gg(Dm / 128, M_ROWS / 128);   // (8, 32)
    const dim3 blk(256);
    const size_t smem = 32768;               // 2 x (8KB A + 8KB B) double buffer

    // --- Q branch ---
    gemm_wmma<1024, 1024, 12, 4096, 1024, 2048><<<gg, blk, smem, stream>>>(
        xb, WqB, Wq_b, Qcat, (float*)nullptr);                         // q_in -> Qcat[:, :D]
    pad_zero<<<(Bsz * 2 * Dm + 255) / 256, 256, 0, stream>>>(Qpad);
    pad_copy<<<(M_ROWS * Dm) / 256, 256, 0, stream>>>(Qcat, Qpad);
    gemm_wmma<3072, 1024, 11, 2050, 1024, 2048><<<gg, blk, smem, stream>>>(
        Qpad, convqR, convq_b, Qcat + Dm, (float*)nullptr);            // conv3 -> Qcat[:, D:]
    gemm_wmma<2048, 2048, 12, 4096, 2048, 1024><<<gg, blk, smem, stream>>>(
        Qcat, qprojB, qproj_b, Qbuf, (float*)nullptr);                 // Q

    // --- K branch ---
    gemm_wmma<1024, 1024, 12, 4096, 1024, 2048><<<gg, blk, smem, stream>>>(
        xb, WkB, Wk_b, Kcat, (float*)nullptr);
    pad_zero<<<(Bsz * 2 * Dm + 255) / 256, 256, 0, stream>>>(Kpad);
    pad_copy<<<(M_ROWS * Dm) / 256, 256, 0, stream>>>(Kcat, Kpad);
    gemm_wmma<3072, 1024, 11, 2050, 1024, 2048><<<gg, blk, smem, stream>>>(
        Kpad, convkR, convk_b, Kcat + Dm, (float*)nullptr);
    gemm_wmma<2048, 2048, 12, 4096, 2048, 1024><<<gg, blk, smem, stream>>>(
        Kcat, kprojB, kproj_b, Kbuf, (float*)nullptr);

    // --- V ---
    gemm_wmma<1024, 1024, 12, 4096, 1024, 1024><<<gg, blk, smem, stream>>>(
        xb, WvB, Wv_b, Vbuf, (float*)nullptr);

    // --- tau / delta ---
    dw_gelu<<<(M_ROWS * 2 * Dm) / 256, 256, 0, stream>>>(x, tau1_w, tau1_b, gbuf);
    proj_sig<<<(M_ROWS * Hn) / 256, 256, 0, stream>>>(gbuf, tau2_w, tau2_b, taubuf);
    dw_gelu<<<(M_ROWS * 2 * Dm) / 256, 256, 0, stream>>>(x, del1_w, del1_b, gbuf);
    proj_sig<<<(M_ROWS * Hn) / 256, 256, 0, stream>>>(gbuf, del2_w, del2_b, delbuf);

    // --- attention ---
    flash_attn<<<dim3(Lseq / 128, Hn, Bsz), blk, 0, stream>>>(
        Qbuf, Kbuf, Vbuf, taubuf, delbuf, attnO);

    // --- output projection (fp32 out) ---
    gemm_wmma<1024, 1024, 12, 4096, 1024, 1024><<<gg, blk, smem, stream>>>(
        attnO, outwB, out_b, (__bf16*)nullptr, out);
}
